// xLSTMCell_76390288327651
// MI455X (gfx1250) — matrix operand order repaired and checked by reading, not compiled, not run on hardware
//
#include <hip/hip_runtime.h>
#include <hip/hip_bf16.h>
#include <stdint.h>

typedef __attribute__((ext_vector_type(16))) __bf16 v16bf;
typedef __attribute__((ext_vector_type(8)))  float  v8f;
typedef __attribute__((ext_vector_type(4)))  unsigned int v4u;
typedef __attribute__((ext_vector_type(4)))  int v4i;
typedef __attribute__((ext_vector_type(4)))  float v4f;

#define B_DIM 16384
#define I_DIM 512
#define H_DIM 512
#define K_DIM 1024      // I + H
#define NGATE 5         // i, f, c, o, e
#define BM 128          // rows per block
#define BN 32           // H-cols per block
#define KB 64           // K per LDS stage (2 WMMA sub-steps of 32)
#define NK (K_DIM / KB) // 16 stages
#define LDSPAD 72       // LDS row stride in bf16 units (144 B, conflict-free)

#define SA_BUF (BM * LDSPAD)           // 9216 bf16 per A buffer
#define SW_BUF (NGATE * BN * LDSPAD)   // 11520 bf16 per W buffer
#define SMEM_BYTES ((2 * SA_BUF + 2 * SW_BUF) * 2)   // 82,944 B (dynamic LDS)

// CDNA5 async global->LDS path (ASYNCcnt-tracked), with safe fallback.
#if defined(__AMDGCN__) && __has_builtin(__builtin_amdgcn_global_load_async_to_lds_b128)
#define USE_ASYNC 1
#else
#define USE_ASYNC 0
#endif

#define AS1 __attribute__((address_space(1)))
#define AS3 __attribute__((address_space(3)))

__device__ __forceinline__ void wait_async_copies() {
#if USE_ASYNC
#if __has_builtin(__builtin_amdgcn_s_wait_asynccnt)
  __builtin_amdgcn_s_wait_asynccnt(0);
#else
  asm volatile("s_wait_asynccnt 0x0" ::: "memory");
#endif
#endif
}

__device__ __forceinline__ void sched_fence() {
#if defined(__AMDGCN__) && __has_builtin(__builtin_amdgcn_sched_barrier)
  __builtin_amdgcn_sched_barrier(0);   // keep ds_loads above, wmmas below
#endif
}

union FragU { v4u q[2]; v16bf v; };

__device__ __forceinline__ unsigned short f2bf(float f) {
  unsigned int u = __builtin_bit_cast(unsigned int, f);
  u += 0x7FFFu + ((u >> 16) & 1u);   // round-to-nearest-even
  return (unsigned short)(u >> 16);
}

__device__ __forceinline__ float sigf(float v) {
  return 1.0f / (1.0f + __expf(-v));
}
__device__ __forceinline__ float tanh_fast(float v) {
  float a = __builtin_fabsf(v);
  float e = __expf(-2.0f * a);
  float t = (1.0f - e) / (1.0f + e);
  return v < 0.0f ? -t : t;
}

// ---------------- pack kernels: f32 -> bf16 staging in workspace ------------

__global__ void pack_A_kernel(const float* __restrict__ x,
                              const float* __restrict__ h,
                              v4u* __restrict__ out) {
  int idx = blockIdx.x * 256 + threadIdx.x;        // one 8-elem bf16 chunk
  int b  = idx >> 7;                               // K/8 = 128 chunks per row
  int kc = idx & 127;
  int k0 = kc << 3;
  const float* src = (k0 < I_DIM) ? (x + (size_t)b * I_DIM + k0)
                                  : (h + (size_t)b * H_DIM + (k0 - I_DIM));
  v4f f0 = *(const v4f*)(src);
  v4f f1 = *(const v4f*)(src + 4);
  v4u o;
  o.x = (unsigned)f2bf(f0.x) | ((unsigned)f2bf(f0.y) << 16);
  o.y = (unsigned)f2bf(f0.z) | ((unsigned)f2bf(f0.w) << 16);
  o.z = (unsigned)f2bf(f1.x) | ((unsigned)f2bf(f1.y) << 16);
  o.w = (unsigned)f2bf(f1.z) | ((unsigned)f2bf(f1.w) << 16);
  out[idx] = o;
}

__global__ void pack_W_kernel(const float* __restrict__ Wxi, const float* __restrict__ Whi,
                              const float* __restrict__ Wxf, const float* __restrict__ Whf,
                              const float* __restrict__ Wxc, const float* __restrict__ Whc,
                              const float* __restrict__ Wxo, const float* __restrict__ Who,
                              const float* __restrict__ We,
                              v4u* __restrict__ out) {
  int idx = blockIdx.x * 256 + threadIdx.x;        // [0, 5*512*128)
  int g   = idx >> 16;                             // 512*128 = 65536 per gate
  int rem = idx & 65535;
  int j   = rem >> 7;
  int k0  = (rem & 127) << 3;
  const float* src;
  if (g == 4) {
    src = We + (size_t)j * K_DIM + k0;
  } else {
    const float* wx; const float* wh;
    switch (g) {
      case 0:  wx = Wxi; wh = Whi; break;
      case 1:  wx = Wxf; wh = Whf; break;
      case 2:  wx = Wxc; wh = Whc; break;
      default: wx = Wxo; wh = Who; break;
    }
    src = (k0 < I_DIM) ? (wx + (size_t)j * I_DIM + k0)
                       : (wh + (size_t)j * H_DIM + (k0 - I_DIM));
  }
  v4f f0 = *(const v4f*)(src);
  v4f f1 = *(const v4f*)(src + 4);
  v4u o;
  o.x = (unsigned)f2bf(f0.x) | ((unsigned)f2bf(f0.y) << 16);
  o.y = (unsigned)f2bf(f0.z) | ((unsigned)f2bf(f0.w) << 16);
  o.z = (unsigned)f2bf(f1.x) | ((unsigned)f2bf(f1.y) << 16);
  o.w = (unsigned)f2bf(f1.z) | ((unsigned)f2bf(f1.w) << 16);
  out[idx] = o;
}

__global__ void pack_bias_kernel(const float* __restrict__ bxi, const float* __restrict__ bhi,
                                 const float* __restrict__ bxf, const float* __restrict__ bhf,
                                 const float* __restrict__ bxc, const float* __restrict__ bhc,
                                 const float* __restrict__ bxo, const float* __restrict__ bho,
                                 const float* __restrict__ be,
                                 float* __restrict__ out) {
  int idx = blockIdx.x * 256 + threadIdx.x;
  if (idx >= NGATE * H_DIM) return;
  int g = idx / H_DIM, j = idx % H_DIM;
  float v;
  switch (g) {
    case 0:  v = bxi[j] + bhi[j]; break;
    case 1:  v = bxf[j] + bhf[j]; break;
    case 2:  v = bxc[j] + bhc[j]; break;
    case 3:  v = bxo[j] + bho[j]; break;
    default: v = be[j]; break;
  }
  out[idx] = v;
}

// ---------------- main fused WMMA kernel ------------------------------------

__launch_bounds__(256, 1)
__global__ void xlstm_main_kernel(const uint16_t* __restrict__ Abf,
                                  const uint16_t* __restrict__ Wp,
                                  const float* __restrict__ biasp,
                                  const float* __restrict__ c_prev,
                                  float* __restrict__ out) {
  extern __shared__ __align__(16) uint16_t smem[];
  uint16_t* sA = smem;                 // [2][BM * LDSPAD]
  uint16_t* sW = smem + 2 * SA_BUF;    // [2][NGATE * BN * LDSPAD]

  const int tid  = threadIdx.x;
  const int wave = tid >> 5;       // 8 waves (wave32)
  const int lane = tid & 31;
  const int half = lane >> 4;
  const int m    = lane & 15;

  const int row0 = blockIdx.x * BM;        // batch rows
  const int col0 = blockIdx.y * BN;        // H columns

  const v4u* gA = (const v4u*)Abf;         // (B x K/8) of v4u
  const v4u* gW = (const v4u*)Wp;          // (5*512 x K/8) of v4u

  // stage one KB-wide K-block into LDS buffer `buf`
  auto copyA = [&](int buf, int kb) {
    int kq = kb * (KB / 8);                            // v4u offset along K
    #pragma unroll
    for (int u = tid; u < (BM * KB / 8); u += 256) {   // 1024 v4u, 4 iters
      int r = u >> 3, c8 = u & 7;
      const v4u* src = &gA[(size_t)(row0 + r) * (K_DIM / 8) + kq + c8];
      uint16_t* dst = &sA[buf * SA_BUF + r * LDSPAD + c8 * 8];
#if USE_ASYNC
      __builtin_amdgcn_global_load_async_to_lds_b128(
          (AS1 v4i*)src, (AS3 v4i*)dst, 0, 0);
#else
      *(v4u*)dst = *src;
#endif
    }
  };
  auto copyW = [&](int buf, int kb) {
    int kq = kb * (KB / 8);
    for (int u = tid; u < (NGATE * BN * KB / 8); u += 256) { // 1280 v4u, 5 iters
      int r = u >> 3, c8 = u & 7;
      int g = r >> 5, jj = r & 31;
      const v4u* src = &gW[(size_t)(g * H_DIM + col0 + jj) * (K_DIM / 8) + kq + c8];
      uint16_t* dst = &sW[buf * SW_BUF + (g * BN + jj) * LDSPAD + c8 * 8];
#if USE_ASYNC
      __builtin_amdgcn_global_load_async_to_lds_b128(
          (AS1 v4i*)src, (AS3 v4i*)dst, 0, 0);
#else
      *(v4u*)dst = *src;
#endif
    }
  };

  v8f acc[NGATE][2] = {};   // 10 x 16x16 f32 tiles = 80 VGPRs

  copyA(0, 0);
  copyW(0, 0);
  wait_async_copies();
  __syncthreads();

  const int arowbase = (wave * 16 + m) * LDSPAD;

  for (int kb = 0; kb < NK; ++kb) {
    const int buf = kb & 1;
    if (kb + 1 < NK) {
      copyA(buf ^ 1, kb + 1);
      copyW(buf ^ 1, kb + 1);
      if (kb + 2 < NK) {
        // speculative L2 prefetch of the K-block after next
        __builtin_prefetch((const char*)&gA[(size_t)(row0 + (tid >> 2)) * (K_DIM / 8)
                                            + (kb + 2) * (KB / 8)], 0, 0);
      }
    }

    #pragma unroll
    for (int ks = 0; ks < 2; ++ks) {      // two 32-K WMMA sub-steps per stage
      const int koff = ks * 32;           // bf16 offset within staged rows

      // A fragment: 16x32 bf16 (lane m rows, K split by lane-half)
      FragU fa;
      fa.q[0] = *(const v4u*)&sA[buf * SA_BUF + arowbase + koff + 8 * half];
      fa.q[1] = *(const v4u*)&sA[buf * SA_BUF + arowbase + koff + 16 + 8 * half];

      // batch ALL B fragments, then fence so ds_loads pipeline ahead of WMMAs
      FragU fb[NGATE][2];
      #pragma unroll
      for (int g = 0; g < NGATE; ++g) {
        #pragma unroll
        for (int nt = 0; nt < 2; ++nt) {
          const int wrow = buf * SW_BUF + (g * BN + nt * 16 + m) * LDSPAD + koff;
          fb[g][nt].q[0] = *(const v4u*)&sW[wrow + 8 * half];
          fb[g][nt].q[1] = *(const v4u*)&sW[wrow + 16 + 8 * half];
        }
      }
      sched_fence();
      #pragma unroll
      for (int g = 0; g < NGATE; ++g) {
        #pragma unroll
        for (int nt = 0; nt < 2; ++nt) {
          acc[g][nt] = __builtin_amdgcn_wmma_f32_16x16x32_bf16(
              false, fa.v, false, fb[g][nt].v, (short)0, acc[g][nt], false, false);
        }
      }
      sched_fence();
    }

    wait_async_copies();   // next-buffer async copies landed in LDS
    __syncthreads();
  }

  // ----- fused elementwise epilogue: gates -> (h, c) -----
  const int rowBase = row0 + wave * 16;
  #pragma unroll
  for (int nt = 0; nt < 2; ++nt) {
    const int j = col0 + nt * 16 + m;
    const float bi = biasp[j];
    const float bf_ = biasp[H_DIM + j];
    const float bc_ = biasp[2 * H_DIM + j];
    const float bo_ = biasp[3 * H_DIM + j];
    const float be_ = biasp[4 * H_DIM + j];
    #pragma unroll
    for (int v = 0; v < 8; ++v) {
      const int row = rowBase + v + 8 * half;   // C/D layout: M = v + 8*(lane>=16)
      const float gi = acc[0][nt][v] + bi;
      const float gf = acc[1][nt][v] + bf_;
      const float gc = acc[2][nt][v] + bc_;
      const float go = acc[3][nt][v] + bo_;
      const float ge = acc[4][nt][v] + be_;
      const float iv = sigf(gi);
      const float fv = sigf(gf);
      const float gv = tanh_fast(gc);
      const float ov = sigf(go);
      const float cp = __builtin_nontemporal_load(&c_prev[(size_t)row * H_DIM + j]);
      const float cn = fv * cp + iv * gv;
      const float ef = __expf(tanh_fast(ge));
      const float hv = ov * tanh_fast(cn) * ef;
      __builtin_nontemporal_store(hv, &out[(size_t)row * H_DIM + j]);                         // h
      __builtin_nontemporal_store(cn, &out[(size_t)B_DIM * H_DIM + (size_t)row * H_DIM + j]); // c
    }
  }
}

// ---------------- host launch ----------------------------------------------

extern "C" void kernel_launch(void* const* d_in, const int* in_sizes, int n_in,
                              void* d_out, int out_size, void* d_ws, size_t ws_size,
                              hipStream_t stream) {
  (void)in_sizes; (void)n_in; (void)out_size; (void)ws_size;
  const float* x      = (const float*)d_in[0];
  const float* h_prev = (const float*)d_in[1];
  const float* c_prev = (const float*)d_in[2];
  const float* Wxi = (const float*)d_in[3];  const float* bxi = (const float*)d_in[4];
  const float* Whi = (const float*)d_in[5];  const float* bhi = (const float*)d_in[6];
  const float* Wxf = (const float*)d_in[7];  const float* bxf = (const float*)d_in[8];
  const float* Whf = (const float*)d_in[9];  const float* bhf = (const float*)d_in[10];
  const float* Wxc = (const float*)d_in[11]; const float* bxc = (const float*)d_in[12];
  const float* Whc = (const float*)d_in[13]; const float* bhc = (const float*)d_in[14];
  const float* Wxo = (const float*)d_in[15]; const float* bxo = (const float*)d_in[16];
  const float* Who = (const float*)d_in[17]; const float* bho = (const float*)d_in[18];
  const float* We  = (const float*)d_in[19]; const float* be  = (const float*)d_in[20];

  // workspace layout (bf16 staging), all 16B aligned
  uint16_t* Abf = (uint16_t*)d_ws;                                   // 33,554,432 B
  uint16_t* Wpk = Abf + (size_t)B_DIM * K_DIM;                       //  5,242,880 B
  float*  biasp = (float*)(Wpk + (size_t)NGATE * H_DIM * K_DIM);     //     10,240 B

  pack_A_kernel<<<(B_DIM * K_DIM / 8) / 256, 256, 0, stream>>>(x, h_prev, (v4u*)Abf);
  pack_W_kernel<<<(NGATE * H_DIM * K_DIM / 8) / 256, 256, 0, stream>>>(
      Wxi, Whi, Wxf, Whf, Wxc, Whc, Wxo, Who, We, (v4u*)Wpk);
  pack_bias_kernel<<<(NGATE * H_DIM + 255) / 256, 256, 0, stream>>>(
      bxi, bhi, bxf, bhf, bxc, bhc, bxo, bho, be, biasp);

  dim3 grid(B_DIM / BM, H_DIM / BN);   // 128 x 16
  xlstm_main_kernel<<<grid, 256, SMEM_BYTES, stream>>>(Abf, Wpk, biasp, c_prev, (float*)d_out);
}
